// CrossAttention_66614942761157
// MI455X (gfx1250) — compile-verified, matrix operand order checked by reading, not run on hardware
//
#include <hip/hip_runtime.h>
#include <hip/hip_bf16.h>

// ---------------------------------------------------------------------------
// CDNA5 (gfx1250) co-attention pipeline, bf16 WMMA everywhere.
//   1. convert fp32 -> bf16 (inputs + W)
//   2. proj GEMM (WMMA):  A = relu(X W^T + b)  -> bf16 [L,H] and [H,L]
//   3. scores GEMM (WMMA): S = A Bf^T, mask -> -1e9, fp32
//   4. row softmax  -> attnB bf16 [LA,LB]
//   5. col softmax (online stats + LDS transpose) -> attnAT bf16 [LB,LA]
//   6. context GEMMs (WMMA): cvB = attnB BfT^T, cvA = attnAT AT^T
// GEMM tile copies use GLOBAL_LOAD_ASYNC_TO_LDS_B128 (ASYNCcnt) with double
// buffering when the builtin exists; falls back to load+ds_store otherwise.
// ---------------------------------------------------------------------------

typedef __attribute__((ext_vector_type(16))) __bf16 v16bf;
typedef __attribute__((ext_vector_type(8)))  float  v8f;

union FragAB { v16bf v; uint4 q[2]; };
union FragC  { v8f   v; float f[8]; };

#if defined(__has_builtin)
#  if __has_builtin(__builtin_amdgcn_global_load_async_to_lds_b128)
#    define HAVE_ASYNC_COPY 1
#  endif
#  if __has_builtin(__builtin_amdgcn_s_wait_asynccnt)
#    define HAVE_WAIT_ASYNC 1
#  endif
#endif

#if defined(HAVE_ASYNC_COPY)
typedef int v4i_vs __attribute__((vector_size(4 * sizeof(int))));
typedef __attribute__((address_space(1))) v4i_vs* g_v4i_ptr;   // global (AS1)
typedef __attribute__((address_space(3))) v4i_vs* l_v4i_ptr;   // LDS (AS3)

__device__ __forceinline__ void async_copy_b128(const unsigned short* g, unsigned short* l) {
    __builtin_amdgcn_global_load_async_to_lds_b128(
        (g_v4i_ptr)g, (l_v4i_ptr)l, 0, 0);
}
#endif

__device__ __forceinline__ void wait_async_le(int n) {
#if defined(HAVE_ASYNC_COPY)
#  if defined(HAVE_WAIT_ASYNC)
    if (n == 0) __builtin_amdgcn_s_wait_asynccnt(0);
    else        __builtin_amdgcn_s_wait_asynccnt(4);
#  else
    if (n == 0) asm volatile("s_wait_asynccnt 0x0" ::: "memory");
    else        asm volatile("s_wait_asynccnt 0x4" ::: "memory");
#  endif
#else
    (void)n;
#endif
}

__device__ __forceinline__ unsigned short f2bf(float f) {
    unsigned u = __float_as_uint(f);
    unsigned r = u + 0x7FFFu + ((u >> 16) & 1u);   // round-to-nearest-even
    return (unsigned short)(r >> 16);
}

__global__ __launch_bounds__(256)
void f32_to_bf16_kernel(const float* __restrict__ in, unsigned short* __restrict__ out, size_t n) {
    size_t i = (size_t)blockIdx.x * blockDim.x + threadIdx.x;
    size_t stride = (size_t)gridDim.x * blockDim.x;
    for (; i < n; i += stride) out[i] = f2bf(in[i]);
}

// ---------------------------------------------------------------------------
// Tiled NT GEMM: C[M,N] = P[M,K] * Q[N,K]^T, operands bf16 (K contiguous),
// f32 accumulate via v_wmma_f32_16x16x32_bf16.
// EPI 0: bias+relu, store bf16 row-major + transposed
// EPI 1: mask -> -1e9, store fp32
// EPI 2: store fp32
// ---------------------------------------------------------------------------
#define BM 128
#define BN 128
#define BK 32
#define LDSS 40   // ushort stride per LDS row (32 + 8 pad), 80B: keeps 16B align

template<int EPI>
__global__ __launch_bounds__(256)
void gemm_nt(const unsigned short* __restrict__ Pg,
             const unsigned short* __restrict__ Qg,
             int M, int N, int K,
             long pBatch, long qBatch,
             float* __restrict__ outF, long outFBatch,
             unsigned short* __restrict__ outBf,  long outBfBatch,
             unsigned short* __restrict__ outBfT, long outBfTBatch,
             const float* __restrict__ bias,
             const float* __restrict__ maskP, const float* __restrict__ maskQ)
{
#if defined(HAVE_ASYNC_COPY)
    __shared__ unsigned short As[2][BM * LDSS];
    __shared__ unsigned short Bs[2][BN * LDSS];
#else
    __shared__ unsigned short As1[BM * LDSS];
    __shared__ unsigned short Bs1[BN * LDSS];
#endif

    const int b   = blockIdx.z;
    const int m0  = blockIdx.y * BM;
    const int n0  = blockIdx.x * BN;
    const int tid = threadIdx.x;
    const int lane  = tid & 31;
    const int wave  = tid >> 5;     // 0..7
    const int wm    = wave & 3;     // 4 wave-rows of 32
    const int wn    = wave >> 2;    // 2 wave-cols of 64
    const int lhalf = lane >> 4;    // 0/1 (lane half per WMMA layout)
    const int lrow  = lane & 15;

    const unsigned short* P = Pg + (size_t)b * pBatch;
    const unsigned short* Q = Qg + (size_t)b * qBatch;

    const int ldRow = tid >> 1;           // 0..127
    const int ldCol = (tid & 1) * 16;     // 0 / 16

    FragC acc[2][4];
    #pragma unroll
    for (int i = 0; i < 2; i++)
        #pragma unroll
        for (int j = 0; j < 4; j++)
            #pragma unroll
            for (int e = 0; e < 8; e++) acc[i][j].f[e] = 0.0f;

    // fragment loads per ISA A/B layout (K split by lane half) + 8 WMMAs
    auto compute_tile = [&](const unsigned short* Asb, const unsigned short* Bsb) {
        FragAB afr[2], bfr[4];
        #pragma unroll
        for (int mi = 0; mi < 2; mi++) {
            int r = wm * 32 + mi * 16 + lrow;
            afr[mi].q[0] = *reinterpret_cast<const uint4*>(&Asb[r * LDSS + lhalf * 8]);
            afr[mi].q[1] = *reinterpret_cast<const uint4*>(&Asb[r * LDSS + 16 + lhalf * 8]);
        }
        #pragma unroll
        for (int ni = 0; ni < 4; ni++) {
            int c = wn * 64 + ni * 16 + lrow;
            bfr[ni].q[0] = *reinterpret_cast<const uint4*>(&Bsb[c * LDSS + lhalf * 8]);
            bfr[ni].q[1] = *reinterpret_cast<const uint4*>(&Bsb[c * LDSS + 16 + lhalf * 8]);
        }
        #pragma unroll
        for (int mi = 0; mi < 2; mi++)
            #pragma unroll
            for (int ni = 0; ni < 4; ni++)
                acc[mi][ni].v = __builtin_amdgcn_wmma_f32_16x16x32_bf16(
                    false, afr[mi].v, false, bfr[ni].v,
                    (short)0, acc[mi][ni].v, false, false);
    };

#if defined(HAVE_ASYNC_COPY)
    // -------- async double-buffered pipeline (4 async b128 per thread/tile) --
    auto issue_copy = [&](int buf, int k0) {
        const unsigned short* ps = P + (size_t)(m0 + ldRow) * K + k0 + ldCol;
        unsigned short* pd = &As[buf][ldRow * LDSS + ldCol];
        async_copy_b128(ps,     pd);
        async_copy_b128(ps + 8, pd + 8);
        const unsigned short* qs = Q + (size_t)(n0 + ldRow) * K + k0 + ldCol;
        unsigned short* qd = &Bs[buf][ldRow * LDSS + ldCol];
        async_copy_b128(qs,     qd);
        async_copy_b128(qs + 8, qd + 8);
    };
    const int nk = K / BK;
    issue_copy(0, 0);
    for (int i = 0; i < nk; i++) {
        const int cur = i & 1;
        if (i + 1 < nk) {
            issue_copy(cur ^ 1, (i + 1) * BK);  // stream next tile during WMMA
            wait_async_le(4);                   // current tile's 4 ops done
        } else {
            wait_async_le(0);
        }
        __syncthreads();
        compute_tile(As[cur], Bs[cur]);
        __syncthreads();                        // buffer free before re-target
    }
#else
    // -------- fallback: load to VGPR + ds_store (compiler pipelines) --------
    for (int k0 = 0; k0 < K; k0 += BK) {
        {
            const uint4* sp = reinterpret_cast<const uint4*>(P + (size_t)(m0 + ldRow) * K + k0 + ldCol);
            uint4 a0 = sp[0], a1 = sp[1];
            const uint4* sq = reinterpret_cast<const uint4*>(Q + (size_t)(n0 + ldRow) * K + k0 + ldCol);
            uint4 b0 = sq[0], b1 = sq[1];
            uint4* dp = reinterpret_cast<uint4*>(&As1[ldRow * LDSS + ldCol]);
            dp[0] = a0; dp[1] = a1;
            uint4* dq = reinterpret_cast<uint4*>(&Bs1[ldRow * LDSS + ldCol]);
            dq[0] = b0; dq[1] = b1;
        }
        __syncthreads();
        compute_tile(As1, Bs1);
        __syncthreads();
    }
#endif

    // ---- epilogue: C VGPR j -> M = j + lhalf*8, N = lrow ----
    #pragma unroll
    for (int mi = 0; mi < 2; mi++) {
        #pragma unroll
        for (int ni = 0; ni < 4; ni++) {
            int cg = n0 + wn * 64 + ni * 16 + lrow;
            #pragma unroll
            for (int j = 0; j < 8; j++) {
                int rg = m0 + wm * 32 + mi * 16 + lhalf * 8 + j;
                float v = acc[mi][ni].f[j];
                if (EPI == 0) {
                    v += bias[cg];
                    v = v > 0.0f ? v : 0.0f;
                    unsigned short h = f2bf(v);
                    outBf [(size_t)b * outBfBatch  + (size_t)rg * N + cg] = h;
                    outBfT[(size_t)b * outBfTBatch + (size_t)cg * M + rg] = h;
                } else if (EPI == 1) {
                    if (maskP[(size_t)b * M + rg] * maskQ[(size_t)b * N + cg] == 0.0f)
                        v = -1.0e9f;
                    outF[(size_t)b * outFBatch + (size_t)rg * N + cg] = v;
                } else {
                    outF[(size_t)b * outFBatch + (size_t)rg * N + cg] = v;
                }
            }
        }
    }
}

// ---------------------------------------------------------------------------
// softmax over last axis (rows); assumes LB == 2048 (8 * 256)
// ---------------------------------------------------------------------------
__global__ __launch_bounds__(256)
void row_softmax(const float* __restrict__ S, unsigned short* __restrict__ attn, int LB) {
    __shared__ float red[256];
    const size_t row = blockIdx.x;
    const float* srow = S + row * (size_t)LB;
    const int t = threadIdx.x;
    float x[8];
    float mx = -3.4e38f;
    #pragma unroll
    for (int i = 0; i < 8; i++) { x[i] = srow[t + i * 256]; mx = fmaxf(mx, x[i]); }
    red[t] = mx; __syncthreads();
    for (int s = 128; s > 0; s >>= 1) { if (t < s) red[t] = fmaxf(red[t], red[t + s]); __syncthreads(); }
    mx = red[0]; __syncthreads();
    float sum = 0.0f;
    #pragma unroll
    for (int i = 0; i < 8; i++) { x[i] = __expf(x[i] - mx); sum += x[i]; }
    red[t] = sum; __syncthreads();
    for (int s = 128; s > 0; s >>= 1) { if (t < s) red[t] += red[t + s]; __syncthreads(); }
    float inv = 1.0f / red[0];
    unsigned short* arow = attn + row * (size_t)LB;
    #pragma unroll
    for (int i = 0; i < 8; i++) arow[t + i * 256] = f2bf(x[i] * inv);
}

// online column max/sum (softmax over axis=1)
__global__ __launch_bounds__(256)
void col_stats(const float* __restrict__ S, float* __restrict__ cmax, float* __restrict__ csum,
               int LA, int LB) {
    int b = blockIdx.y;
    int m = blockIdx.x * 256 + threadIdx.x;
    const float* Sb = S + (size_t)b * LA * LB;
    float mx = -3.4e38f, s = 0.0f;
    for (int l = 0; l < LA; l++) {
        float x = Sb[(size_t)l * LB + m];
        float nm = fmaxf(mx, x);
        s = s * __expf(mx - nm) + __expf(x - nm);
        mx = nm;
    }
    cmax[(size_t)b * LB + m] = mx;
    csum[(size_t)b * LB + m] = s;
}

// normalize by column stats and write transposed bf16 [LB, LA] via LDS tile
__global__ __launch_bounds__(256)
void col_norm_T(const float* __restrict__ S, const float* __restrict__ cmax,
                const float* __restrict__ csum, unsigned short* __restrict__ attnT,
                int LA, int LB) {
    __shared__ float tile[32][33];
    int b = blockIdx.z;
    int l0 = blockIdx.x * 32, m0 = blockIdx.y * 32;
    int tx = threadIdx.x & 31, ty = threadIdx.x >> 5;   // ty 0..7
    const float* Sb = S + (size_t)b * LA * LB;
    #pragma unroll
    for (int r = 0; r < 4; r++) {
        int l = l0 + ty + r * 8, m = m0 + tx;
        float x = Sb[(size_t)l * LB + m];
        tile[ty + r * 8][tx] = __expf(x - cmax[(size_t)b * LB + m]) / csum[(size_t)b * LB + m];
    }
    __syncthreads();
    unsigned short* T = attnT + (size_t)b * LB * LA;
    #pragma unroll
    for (int r = 0; r < 4; r++) {
        int m = m0 + ty + r * 8, l = l0 + tx;
        T[(size_t)m * LA + l] = f2bf(tile[tx][ty + r * 8]);
    }
}

// ---------------------------------------------------------------------------
extern "C" void kernel_launch(void* const* d_in, const int* in_sizes, int n_in,
                              void* d_out, int out_size, void* d_ws, size_t ws_size,
                              hipStream_t stream) {
    (void)in_sizes; (void)n_in; (void)out_size; (void)ws_size;
    const int B = 8, LA = 2048, LB = 2048, D = 512, H = 512;

    const float* inputA = (const float*)d_in[0];
    const float* inputB = (const float*)d_in[1];
    const float* maskA  = (const float*)d_in[2];
    const float* maskB  = (const float*)d_in[3];
    const float* W      = (const float*)d_in[4];
    const float* bias   = (const float*)d_in[5];
    float* out = (float*)d_out;

    const size_t PROJ = (size_t)B * LA * H;     // = B*L*D here too
    const size_t SC   = (size_t)B * LA * LB;

    char* w = (char*)d_ws;
    unsigned short* inA_bf = (unsigned short*)w; w += PROJ * 2;
    unsigned short* inB_bf = (unsigned short*)w; w += PROJ * 2;
    unsigned short* W_bf   = (unsigned short*)w; w += (size_t)H * D * 2;
    unsigned short* A_bf   = (unsigned short*)w; w += PROJ * 2;   // [B,LA,H]
    unsigned short* AT_bf  = (unsigned short*)w; w += PROJ * 2;   // [B,H,LA]
    unsigned short* Bf_bf  = (unsigned short*)w; w += PROJ * 2;   // [B,LB,H]
    unsigned short* BfT_bf = (unsigned short*)w; w += PROJ * 2;   // [B,H,LB]
    float*          scores = (float*)w;          w += SC * 4;     // [B,LA,LB]
    unsigned short* attnB  = (unsigned short*)w; w += SC * 2;     // [B,LA,LB]
    unsigned short* attnAT = (unsigned short*)w; w += SC * 2;     // [B,LB,LA]
    float*          cmax   = (float*)w;          w += (size_t)B * LB * 4;
    float*          csum   = (float*)w;          w += (size_t)B * LB * 4;

    // 1. fp32 -> bf16 converts
    f32_to_bf16_kernel<<<4096, 256, 0, stream>>>(inputA, inA_bf, (size_t)B * LA * D);
    f32_to_bf16_kernel<<<4096, 256, 0, stream>>>(inputB, inB_bf, (size_t)B * LB * D);
    f32_to_bf16_kernel<<<256, 256, 0, stream>>>(W, W_bf, (size_t)H * D);

    // 2. projections: relu(X W^T + b)
    dim3 gP(H / BN, LA / BM, B);
    gemm_nt<0><<<gP, 256, 0, stream>>>(inA_bf, W_bf, LA, H, D, (long)LA * D, 0L,
        nullptr, 0L, A_bf, (long)LA * H, AT_bf, (long)H * LA, bias, nullptr, nullptr);
    gemm_nt<0><<<gP, 256, 0, stream>>>(inB_bf, W_bf, LB, H, D, (long)LB * D, 0L,
        nullptr, 0L, Bf_bf, (long)LB * H, BfT_bf, (long)H * LB, bias, nullptr, nullptr);

    // 3. scores = A Bf^T with mask
    dim3 gS(LB / BN, LA / BM, B);
    gemm_nt<1><<<gS, 256, 0, stream>>>(A_bf, Bf_bf, LA, LB, H, (long)LA * H, (long)LB * H,
        scores, (long)LA * LB, nullptr, 0L, nullptr, 0L, nullptr, maskA, maskB);

    // 4. softmax over LB (rows) -> attnB
    row_softmax<<<B * LA, 256, 0, stream>>>(scores, attnB, LB);

    // 5. softmax over LA (cols) -> attnAT (transposed)
    dim3 gC(LB / 256, B);
    col_stats<<<gC, 256, 0, stream>>>(scores, cmax, csum, LA, LB);
    dim3 gT(LA / 32, LB / 32, B);
    col_norm_T<<<gT, 256, 0, stream>>>(scores, cmax, csum, attnAT, LA, LB);

    // 6. context vectors
    // cvB[l,h] = sum_m attnB[l,m] * Bf[m,h] ; Q = BfT [H,LB]
    dim3 gV(H / BN, LA / BM, B);
    gemm_nt<2><<<gV, 256, 0, stream>>>(attnB, BfT_bf, LA, H, LB, (long)LA * LB, (long)H * LB,
        out + (size_t)B * LB * H, (long)LA * H,
        nullptr, 0L, nullptr, 0L, nullptr, nullptr, nullptr);
    // cvA[m,h] = sum_l attnA[l,m] * A[l,h] ; P = attnAT [LB,LA], Q = AT [H,LA]
    dim3 gU(H / BN, LB / BM, B);
    gemm_nt<2><<<gU, 256, 0, stream>>>(attnAT, AT_bf, LB, H, LA, (long)LB * LA, (long)H * LA,
        out, (long)LB * H,
        nullptr, 0L, nullptr, 0L, nullptr, nullptr, nullptr);
}
